// Transformer_4045859193013
// MI455X (gfx1250) — compile-verified, matrix operand order checked by reading, not run on hardware
//
#include <hip/hip_runtime.h>
#include <cstdint>
#include <cstddef>

// ---------------------------------------------------------------------------
// Problem constants (match reference)
// ---------------------------------------------------------------------------
#define B_   4
#define S_   2048
#define D_   1024
#define H_   16
#define K_   64
#define L_   4
#define MROW (B_ * S_)          // 8192 token rows
#define LN_EPS 1e-5f

typedef __attribute__((ext_vector_type(16))) __bf16 v16bf;
typedef __attribute__((ext_vector_type(8)))  float  v8f;
typedef __attribute__((ext_vector_type(4)))  int    v4i;
typedef unsigned short u16;

// ---------------------------------------------------------------------------
// CDNA5 async global->LDS path (ASYNCcnt).  Falls back to sync staging if the
// toolchain does not expose the builtins.
// ---------------------------------------------------------------------------
#if __has_builtin(__builtin_amdgcn_global_load_async_to_lds_b128) && \
    __has_builtin(__builtin_amdgcn_s_wait_asynccnt)
#define ASYNC_LDS 1
#else
#define ASYNC_LDS 0
#endif

#if ASYNC_LDS
static __device__ __forceinline__ void async_cp16(const u16* g, u16* l) {
  // probe-derived clang signature:
  //   (v4i* src, v4i addrspace(3)* dst, Imm int offset, Imm int cpol)
  __builtin_amdgcn_global_load_async_to_lds_b128(
      (v4i*)const_cast<u16*>(g),
      (__attribute__((address_space(3))) v4i*)l, 0, 0);
}
#endif

// ---------------------------------------------------------------------------
// bf16 helpers (round-to-nearest-even)
// ---------------------------------------------------------------------------
static __device__ __forceinline__ u16 f2bf(float f) {
  union { float f; unsigned int u; } v; v.f = f;
  unsigned int r = v.u + 0x7FFFu + ((v.u >> 16) & 1u);
  return (u16)(r >> 16);
}

union FB { v16bf v; unsigned int u[8]; };

// Load a 16x32 bf16 A/B fragment (ISA 16-bit A-matrix layout, wave32):
//  lanes 0-15  : row = lane,      halves 0..7 -> K 0..7,  halves 8..15 -> K 16..23
//  lanes 16-31 : row = lane-16,   halves 0..7 -> K 8..15, halves 8..15 -> K 24..31
static __device__ __forceinline__ v16bf ld_frag(const u16* __restrict__ base,
                                                int strideHalves, int kbase) {
  const int lane = threadIdx.x & 31;
  const unsigned int* p =
      (const unsigned int*)(base + (size_t)(lane & 15) * strideHalves + kbase);
  const int kh = (lane >> 4) << 2;   // dword offset 0 or 4
  FB f;
#pragma unroll
  for (int j = 0; j < 4; ++j) {
    f.u[j]     = p[kh + j];          // K = khalf .. khalf+7
    f.u[4 + j] = p[8 + kh + j];      // K = 16+khalf .. 16+khalf+7
  }
  return f.v;
}

static __device__ __forceinline__ v8f vzero8() {
  v8f z;
#pragma unroll
  for (int i = 0; i < 8; ++i) z[i] = 0.f;
  return z;
}

static __device__ __forceinline__ v8f wmma_bf16(v16bf a, v16bf b, v8f c) {
  return __builtin_amdgcn_wmma_f32_16x16x32_bf16(false, a, false, b,
                                                 (short)0, c, false, false);
}

// ---------------------------------------------------------------------------
// Weight convert + transpose: Wt[n][k] = bf16(W[k][n]),  D x D, tile 32x32
// ---------------------------------------------------------------------------
__global__ __launch_bounds__(256) void k_wcvt(const float* __restrict__ W,
                                              u16* __restrict__ Wt) {
  __shared__ float tile[32][33];
  const int bx = blockIdx.x * 32;        // n tile
  const int by = blockIdx.y * 32;        // k tile
  const int tx = threadIdx.x & 31;
  const int ty = threadIdx.x >> 5;       // 0..7
#pragma unroll
  for (int i = 0; i < 32; i += 8)
    tile[ty + i][tx] = W[(size_t)(by + ty + i) * D_ + bx + tx];
  __syncthreads();
#pragma unroll
  for (int i = 0; i < 32; i += 8)
    Wt[(size_t)(bx + ty + i) * D_ + by + tx] = f2bf(tile[tx][ty + i]);
}

// ---------------------------------------------------------------------------
// Fused LayerNorm + ReLU -> bf16 activation.  One block (256 thr) per row.
// ---------------------------------------------------------------------------
__global__ __launch_bounds__(256) void k_ln_act(const float* __restrict__ x,
                                                const float* __restrict__ g,
                                                const float* __restrict__ bta,
                                                u16* __restrict__ out) {
  __shared__ float red[2][8];
  const int row = blockIdx.x;
  const int t = threadIdx.x;
  const float* xr = x + (size_t)row * D_;
  float v[4], s = 0.f, s2 = 0.f;
#pragma unroll
  for (int i = 0; i < 4; ++i) {
    v[i] = xr[t + 256 * i];
    s  += v[i];
    s2 += v[i] * v[i];
  }
#pragma unroll
  for (int o = 16; o > 0; o >>= 1) {
    s  += __shfl_xor(s,  o);
    s2 += __shfl_xor(s2, o);
  }
  const int wave = t >> 5, lane = t & 31;
  if (lane == 0) { red[0][wave] = s; red[1][wave] = s2; }
  __syncthreads();
  float ts = 0.f, ts2 = 0.f;
#pragma unroll
  for (int i = 0; i < 8; ++i) { ts += red[0][i]; ts2 += red[1][i]; }
  const float mu = ts * (1.f / D_);
  const float var = ts2 * (1.f / D_) - mu * mu;
  const float rs = rsqrtf(var + LN_EPS);
  u16* orow = out + (size_t)row * D_;
#pragma unroll
  for (int i = 0; i < 4; ++i) {
    const int idx = t + 256 * i;
    float y = (v[i] - mu) * rs * g[idx] + bta[idx];
    orow[idx] = f2bf(fmaxf(y, 0.f));
  }
}

// ---------------------------------------------------------------------------
// WMMA GEMM:  C[M x 1024] = A(bf16, M x 1024) * Wt(bf16, 1024 x 1024)^T
//             + bias, epilogue: (acc+bias)*alpha [, *maskrow] [, +resid]
// Tile 128x128x64, 256 threads = 8 waves (4 x 2), wave tile 32x64.
// Double-buffered LDS; staged with GLOBAL_LOAD_ASYNC_TO_LDS_B128 when
// available so stage k+1's DMA overlaps stage k's WMMAs.
// ---------------------------------------------------------------------------
template <bool OUTBF, bool RES, bool MASKZ>
__global__ __launch_bounds__(256) void k_gemm(const u16* __restrict__ A,
                                              const u16* __restrict__ Wt,
                                              const float* __restrict__ bias,
                                              float alpha,
                                              const float* __restrict__ resid,
                                              const unsigned char* __restrict__ mask,
                                              void* __restrict__ outp) {
  __shared__ u16 As[2][128][72];   // pitch 144B: 16B-aligned, bank-spread
  __shared__ u16 Bs[2][128][72];
  const int tid  = threadIdx.x;
  const int wave = tid >> 5;
  const int lane = tid & 31;
  const int wm = wave & 3;             // 0..3 -> 32-row slabs
  const int wn = wave >> 2;            // 0..1 -> 64-col slabs
  const int mBlk = blockIdx.y * 128;
  const int nBlk = blockIdx.x * 128;

  v8f acc[2][4];
#pragma unroll
  for (int i = 0; i < 2; ++i)
#pragma unroll
    for (int j = 0; j < 4; ++j) acc[i][j] = vzero8();

  // stage a 128x64 bf16 tile of A and of Wt into LDS buffer `buf`
  auto stage = [&](int buf, int kk) {
#pragma unroll
    for (int c = tid; c < 1024; c += 256) {       // 4 chunks per matrix
      const int row = c >> 3;
      const int col8 = (c & 7) * 8;
      const u16* ga = A  + (size_t)(mBlk + row) * D_ + kk + col8;
      const u16* gb = Wt + (size_t)(nBlk + row) * D_ + kk + col8;
#if ASYNC_LDS
      async_cp16(ga, &As[buf][row][col8]);
      async_cp16(gb, &Bs[buf][row][col8]);
#else
      *(uint4*)&As[buf][row][col8] = *(const uint4*)ga;
      *(uint4*)&Bs[buf][row][col8] = *(const uint4*)gb;
#endif
    }
  };

  stage(0, 0);
  for (int kb = 0; kb < D_ / 64; ++kb) {
    const int cur = kb & 1;
    const bool more = (kb + 1) < (D_ / 64);
    if (more) stage(cur ^ 1, (kb + 1) * 64);      // prefetch next stage
#if ASYNC_LDS
    if (more) __builtin_amdgcn_s_wait_asynccnt(8);  // only next stage in flight
    else      __builtin_amdgcn_s_wait_asynccnt(0);
#endif
    __syncthreads();                               // publish current stage

    v16bf af[2][2], bf[4][2];
#pragma unroll
    for (int i = 0; i < 2; ++i)
#pragma unroll
      for (int ks = 0; ks < 2; ++ks)
        af[i][ks] = ld_frag(&As[cur][wm * 32 + 16 * i][0], 72, ks * 32);
#pragma unroll
    for (int j = 0; j < 4; ++j)
#pragma unroll
      for (int ks = 0; ks < 2; ++ks)
        bf[j][ks] = ld_frag(&Bs[cur][wn * 64 + 16 * j][0], 72, ks * 32);
#pragma unroll
    for (int ks = 0; ks < 2; ++ks)
#pragma unroll
      for (int i = 0; i < 2; ++i)
#pragma unroll
        for (int j = 0; j < 4; ++j)
          acc[i][j] = wmma_bf16(af[i][ks], bf[j][ks], acc[i][j]);
    __syncthreads();                               // done reading `cur`
  }

  // ---- epilogue: C/D layout -> lane = col (n), VGPR r = row ----
  const int halfbase = (lane >> 4) << 3;
#pragma unroll
  for (int j = 0; j < 4; ++j) {
    const int n = nBlk + wn * 64 + 16 * j + (lane & 15);
    const float bv = bias[n];
#pragma unroll
    for (int i = 0; i < 2; ++i) {
      const int mbase = mBlk + wm * 32 + 16 * i + halfbase;
#pragma unroll
      for (int r = 0; r < 8; ++r) {
        const int m = mbase + r;
        float val = (acc[i][j][r] + bv) * alpha;
        if (MASKZ) val = mask[m] ? val : 0.f;
        if (RES)   val += resid[(size_t)m * D_ + n];
        if (OUTBF) ((u16*)outp)[(size_t)m * D_ + n] = f2bf(val);
        else       ((float*)outp)[(size_t)m * D_ + n] = val;
      }
    }
  }
}

// ---------------------------------------------------------------------------
// Flash attention: block = 128 thr = 4 waves; one (b,h), 64 query rows.
// Each wave owns a 16-row Q tile.  K/V tiles of 32 keys staged in LDS and
// shared by all 4 waves; K staged via async DMA, V transposed through VGPRs
// (overlapping the in-flight K copy).  1/sqrt(K) already folded into Q.
// ---------------------------------------------------------------------------
__global__ __launch_bounds__(128) void k_attn(const u16* __restrict__ Q,
                                              const u16* __restrict__ Kb,
                                              const u16* __restrict__ Vb,
                                              const unsigned char* __restrict__ mask,
                                              u16* __restrict__ O) {
  __shared__ u16  Ks[32][72];     // [key][dim]      pitch 144B
  __shared__ u16  Vs[64][40];     // [vdim][key]     pitch 80B (transposed)
  __shared__ u16  Ps[4][16][40];  // per-wave P (16q x 32key) relayout buffer
  __shared__ float sbias[32];     // additive key mask (0 / -1e30)

  const int blk = blockIdx.x;
  const int qt = blk & 31;
  const int h  = (blk >> 5) & 15;
  const int b  = blk >> 9;
  const int wave = threadIdx.x >> 5;
  const int lane = threadIdx.x & 31;
  const int qrow0 = qt * 64 + wave * 16;
  const int halfbase = (lane >> 4) << 3;

  // Q fragments straight from global (rows are K-contiguous, pitch H*K)
  const u16* qbase = Q + (((size_t)b * S_ + qrow0) * H_ + h) * K_;
  const v16bf q0 = ld_frag(qbase, H_ * K_, 0);
  const v16bf q1 = ld_frag(qbase, H_ * K_, 32);

  v8f o[4];
  float rowm[8], rowl[8];
#pragma unroll
  for (int j = 0; j < 4; ++j) o[j] = vzero8();
#pragma unroll
  for (int r = 0; r < 8; ++r) { rowm[r] = -1e30f; rowl[r] = 0.f; }

  for (int t0 = 0; t0 < S_; t0 += 32) {
    __syncthreads();     // previous block fully consumed before restage
    // ---- stage K tile (row-major): async DMA straight into LDS ----
#pragma unroll
    for (int c = threadIdx.x; c < 256; c += 128) {
      const int row = c >> 3, col8 = (c & 7) * 8;
      const u16* gk = Kb + (((size_t)b * S_ + t0 + row) * H_ + h) * K_ + col8;
#if ASYNC_LDS
      async_cp16(gk, &Ks[row][col8]);
#else
      *(uint4*)&Ks[row][col8] = *(const uint4*)gk;
#endif
    }
    // ---- stage V tile transposed (VALU work overlaps K DMA) ----
#pragma unroll
    for (int c = threadIdx.x; c < 256; c += 128) {
      const int row = c >> 3, col8 = (c & 7) * 8;
      uint4 d = *(const uint4*)(Vb + (((size_t)b * S_ + t0 + row) * H_ + h) * K_ + col8);
      const u16* d16 = (const u16*)&d;
#pragma unroll
      for (int e = 0; e < 8; ++e) Vs[col8 + e][row] = d16[e];
    }
    if (threadIdx.x < 32)
      sbias[threadIdx.x] = mask[(size_t)b * S_ + t0 + threadIdx.x] ? 0.f : -1e30f;
#if ASYNC_LDS
    __builtin_amdgcn_s_wait_asynccnt(0);
#endif
    __syncthreads();

    // ---- S = Q * K^T for two 16-key subtiles ----
    v8f s[2];
#pragma unroll
    for (int kt = 0; kt < 2; ++kt) {
      v8f sf = vzero8();
      const v16bf b0 = ld_frag(&Ks[kt * 16][0], 72, 0);
      const v16bf b1 = ld_frag(&Ks[kt * 16][0], 72, 32);
      sf = wmma_bf16(q0, b0, sf);
      sf = wmma_bf16(q1, b1, sf);
      const float mb = sbias[kt * 16 + (lane & 15)];
#pragma unroll
      for (int r = 0; r < 8; ++r) sf[r] += mb;
      s[kt] = sf;
    }

    // ---- online softmax (rows live across 16-lane halves) ----
#pragma unroll
    for (int r = 0; r < 8; ++r) {
      float mx = fmaxf(s[0][r], s[1][r]);
#pragma unroll
      for (int off = 8; off > 0; off >>= 1) mx = fmaxf(mx, __shfl_xor(mx, off));
      const float mold = rowm[r];
      const float mnew = fmaxf(mold, mx);
      rowm[r] = mnew;
      const float sc = __expf(mold - mnew);
      rowl[r] *= sc;
#pragma unroll
      for (int j = 0; j < 4; ++j) o[j][r] *= sc;
      const float e0 = __expf(s[0][r] - mnew);
      const float e1 = __expf(s[1][r] - mnew);
      float es = e0 + e1;
#pragma unroll
      for (int off = 8; off > 0; off >>= 1) es += __shfl_xor(es, off);
      rowl[r] += es;
      s[0][r] = e0;
      s[1][r] = e1;
    }

    // ---- relayout P (C-frag -> A-frag) through wave-private LDS ----
#pragma unroll
    for (int kt = 0; kt < 2; ++kt)
#pragma unroll
      for (int r = 0; r < 8; ++r)
        Ps[wave][halfbase + r][kt * 16 + (lane & 15)] = f2bf(s[kt][r]);

    const v16bf pa = ld_frag(&Ps[wave][0][0], 40, 0);
#pragma unroll
    for (int j = 0; j < 4; ++j) {
      const v16bf bv = ld_frag(&Vs[16 * j][0], 40, 0);
      o[j] = wmma_bf16(pa, bv, o[j]);
    }
  }

  // ---- normalize and write (B,S,H*V) bf16 ----
#pragma unroll
  for (int r = 0; r < 8; ++r) {
    const float inv = 1.f / rowl[r];
    const size_t srow = (size_t)b * S_ + qrow0 + halfbase + r;
#pragma unroll
    for (int j = 0; j < 4; ++j)
      O[(srow * H_ + h) * K_ + 16 * j + (lane & 15)] = f2bf(o[j][r] * inv);
  }
}

// ---------------------------------------------------------------------------
// Final masked copy to d_out (fp32)
// ---------------------------------------------------------------------------
__global__ __launch_bounds__(256) void k_maskout(const float* __restrict__ X,
                                                 const unsigned char* __restrict__ mask,
                                                 float* __restrict__ out) {
  const size_t i = (size_t)blockIdx.x * 256 + threadIdx.x;
  out[i] = mask[i >> 10] ? X[i] : 0.f;
}

// ---------------------------------------------------------------------------
// Host orchestration
// ---------------------------------------------------------------------------
extern "C" void kernel_launch(void* const* d_in, const int* in_sizes, int n_in,
                              void* d_out, int out_size, void* d_ws, size_t ws_size,
                              hipStream_t stream) {
  (void)in_sizes; (void)n_in; (void)out_size; (void)ws_size;

  const float* x         = (const float*)d_in[0];
  const unsigned char* mask = (const unsigned char*)d_in[1];
  const float* rbb_ln_g  = (const float*)d_in[2];
  const float* rbb_ln_b  = (const float*)d_in[3];
  const float* rbb_W     = (const float*)d_in[4];
  const float* rbb_b     = (const float*)d_in[5];
  const float* rba_ln_g  = (const float*)d_in[6];
  const float* rba_ln_b  = (const float*)d_in[7];
  const float* rba_W     = (const float*)d_in[8];
  const float* rba_b     = (const float*)d_in[9];
  const float* attn_ln_g = (const float*)d_in[10];
  const float* attn_ln_b = (const float*)d_in[11];
  const float* Wq = (const float*)d_in[12];
  const float* bq = (const float*)d_in[13];
  const float* Wk = (const float*)d_in[14];
  const float* bk = (const float*)d_in[15];
  const float* Wv = (const float*)d_in[16];
  const float* bv = (const float*)d_in[17];
  const float* Wo = (const float*)d_in[18];
  const float* bo = (const float*)d_in[19];

  // ---- workspace carve ----
  size_t off = 0;
  auto take = [&](size_t bytes) {
    void* r = (char*)d_ws + off;
    off += (bytes + 255) & ~(size_t)255;
    return r;
  };
  const size_t WELEM = (size_t)D_ * D_;            // 1M
  u16*   WT  = (u16*)  take(24 * WELEM * sizeof(u16));   // 24 transposed bf16 weights
  float* X   = (float*)take((size_t)MROW * D_ * sizeof(float));
  u16*   ABF = (u16*)  take((size_t)MROW * D_ * sizeof(u16));
  float* HF  = (float*)take((size_t)MROW * D_ * sizeof(float));
  u16*   QB  = (u16*)  take((size_t)MROW * D_ * sizeof(u16));
  u16*   KB  = (u16*)  take((size_t)MROW * D_ * sizeof(u16));
  u16*   VB  = (u16*)  take((size_t)MROW * D_ * sizeof(u16));
  u16*   ATT = (u16*)  take((size_t)MROW * D_ * sizeof(u16));

  const dim3 gW(D_ / 32, D_ / 32);
  const dim3 gG(D_ / 128, MROW / 128);             // (8, 64)

  // ---- residual stream init ----
  (void)hipMemcpyAsync(X, x, (size_t)MROW * D_ * sizeof(float),
                       hipMemcpyDeviceToDevice, stream);

  // ---- one-time weight convert+transpose to bf16 ----
  for (int i = 0; i < 4; ++i)
    k_wcvt<<<gW, 256, 0, stream>>>(rbb_W + (size_t)i * WELEM, WT + (size_t)i * WELEM);
  for (int i = 0; i < 4; ++i)
    k_wcvt<<<gW, 256, 0, stream>>>(rba_W + (size_t)i * WELEM, WT + (size_t)(4 + i) * WELEM);
  for (int l = 0; l < L_; ++l) {
    k_wcvt<<<gW, 256, 0, stream>>>(Wq + (size_t)l * WELEM, WT + (size_t)(8 + 4 * l + 0) * WELEM);
    k_wcvt<<<gW, 256, 0, stream>>>(Wk + (size_t)l * WELEM, WT + (size_t)(8 + 4 * l + 1) * WELEM);
    k_wcvt<<<gW, 256, 0, stream>>>(Wv + (size_t)l * WELEM, WT + (size_t)(8 + 4 * l + 2) * WELEM);
    k_wcvt<<<gW, 256, 0, stream>>>(Wo + (size_t)l * WELEM, WT + (size_t)(8 + 4 * l + 3) * WELEM);
  }

  auto resnet = [&](const float* lng, const float* lnb, const u16* wslot,
                    const float* bias) {
    for (int i = 0; i < 2; ++i) {      // NB = 2 blocks
      k_ln_act<<<MROW, 256, 0, stream>>>(X, lng + (size_t)(i * 2 + 0) * D_,
                                         lnb + (size_t)(i * 2 + 0) * D_, ABF);
      k_gemm<false, false, false><<<gG, 256, 0, stream>>>(
          ABF, wslot + (size_t)(i * 2 + 0) * WELEM,
          bias + (size_t)(i * 2 + 0) * D_, 1.f, nullptr, nullptr, HF);
      k_ln_act<<<MROW, 256, 0, stream>>>(HF, lng + (size_t)(i * 2 + 1) * D_,
                                         lnb + (size_t)(i * 2 + 1) * D_, ABF);
      k_gemm<false, true, false><<<gG, 256, 0, stream>>>(
          ABF, wslot + (size_t)(i * 2 + 1) * WELEM,
          bias + (size_t)(i * 2 + 1) * D_, 1.f, X, nullptr, X);
    }
  };

  // ---- resnet before ----
  resnet(rbb_ln_g, rbb_ln_b, WT, rbb_b);

  // ---- attention layers ----
  const float denom = 0.125f;          // 1/sqrt(K=64), folded into Q
  for (int l = 0; l < L_; ++l) {
    const u16* wq = WT + (size_t)(8 + 4 * l + 0) * WELEM;
    const u16* wk = WT + (size_t)(8 + 4 * l + 1) * WELEM;
    const u16* wv = WT + (size_t)(8 + 4 * l + 2) * WELEM;
    const u16* wo = WT + (size_t)(8 + 4 * l + 3) * WELEM;
    k_ln_act<<<MROW, 256, 0, stream>>>(X, attn_ln_g + (size_t)l * D_,
                                       attn_ln_b + (size_t)l * D_, ABF);
    k_gemm<true, false, false><<<gG, 256, 0, stream>>>(
        ABF, wq, bq + (size_t)l * D_, denom, nullptr, nullptr, QB);
    k_gemm<true, false, false><<<gG, 256, 0, stream>>>(
        ABF, wk, bk + (size_t)l * D_, 1.f, nullptr, nullptr, KB);
    k_gemm<true, false, false><<<gG, 256, 0, stream>>>(
        ABF, wv, bv + (size_t)l * D_, 1.f, nullptr, nullptr, VB);
    k_attn<<<B_ * H_ * (S_ / 64), 128, 0, stream>>>(QB, KB, VB, mask, ATT);
    k_gemm<false, true, true><<<gG, 256, 0, stream>>>(
        ATT, wo, bo + (size_t)l * D_, 1.f, X, mask, X);
  }

  // ---- resnet after ----
  resnet(rba_ln_g, rba_ln_b, WT + 4 * WELEM, rba_b);

  // ---- final masked output ----
  k_maskout<<<(MROW * D_) / 256, 256, 0, stream>>>(X, mask, (float*)d_out);
}
